// VFN_InvariantPointAttention_27118423507342
// MI455X (gfx1250) — matrix operand order repaired and checked by reading, not compiled, not run on hardware
//
#include <hip/hip_runtime.h>
#include <math.h>

// ---------------- problem constants ----------------
#define BB   2
#define NN   512
#define CSZ  384
#define HH   12
#define CC   16
#define PPT  8
#define PVV  8
#define KG   16
#define GG   64
#define TOK  (BB*NN)          // 1024
#define INF_ 100000.0f
#define EPS_ 1e-8f
#define SCL_ 0.1f

typedef _Float16 half8 __attribute__((ext_vector_type(8)));
typedef _Float16 v16h  __attribute__((ext_vector_type(16)));
typedef float    v8f   __attribute__((ext_vector_type(8)));

union ABf { v16h v; half8 h[2]; };

// A fragment (16x32 f16): lane row = lane&15 ; K halves: (lane>>4)*8 + [0,8) and 16+(lane>>4)*8+[0,8)
__device__ __forceinline__ v16h load_a_frag(const _Float16* base, int ld, int row, int k0, int lane) {
  ABf u;
  const _Float16* p = base + (size_t)row * ld + k0 + ((lane >> 4) << 3);
  u.h[0] = *(const half8*)p;
  u.h[1] = *(const half8*)(p + 16);
  return u.v;
}
// B fragment (32x16 f16) from transposed B (N x K row-major): lane col = lane&15,
// K halves: (lane>>4)*8 + [0,8) and 16+(lane>>4)*8+[0,8)
__device__ __forceinline__ v16h load_b_frag(const _Float16* baseT, int ld, int col, int k0, int lane) {
  ABf u;
  const _Float16* p = baseT + (size_t)col * ld + k0 + ((lane >> 4) << 3);
  u.h[0] = *(const half8*)p;
  u.h[1] = *(const half8*)(p + 16);
  return u.v;
}

// ---------------- generic WMMA GEMM: C(MxN) = A(MxK) * B(KxN) + bias --------------
// A f16 row-major, BT f16 (N x K) row-major, C fp32 row-major. M%16==0, N%16==0, K%32==0.
__global__ __launch_bounds__(256) void gemm_bias_f16(
    const _Float16* __restrict__ A, const _Float16* __restrict__ BT,
    const float* __restrict__ bias, float* __restrict__ Cout,
    int M, int Nc, int Kd)
{
  int lane = threadIdx.x & 31;
  int wid  = blockIdx.x * 8 + (threadIdx.x >> 5);
  int ntiles = Nc >> 4, mtiles = M >> 4;
  if (wid >= ntiles * mtiles) return;
  int mt = wid / ntiles, nt = wid - mt * ntiles;
  int row = (mt << 4) + (lane & 15);
  int col = (nt << 4) + (lane & 15);
  v8f acc = {};
  for (int k0 = 0; k0 < Kd; k0 += 32) {
    v16h a = load_a_frag(A,  Kd, row, k0, lane);
    v16h b = load_b_frag(BT, Kd, col, k0, lane);
    acc = __builtin_amdgcn_wmma_f32_16x16x32_f16(false, a, false, b, (short)0, acc, false, false);
  }
  float bv = bias ? bias[col] : 0.f;
  int half_ = lane >> 4;
#pragma unroll
  for (int r = 0; r < 8; ++r) {
    int m = (mt << 4) + r + (half_ << 3);
    Cout[(size_t)m * Nc + col] = acc[r] + bv;
  }
}

// ---------------- weight transpose fp32(K x Nsrc) -> f16 (Npad x K) ----------------
__global__ void transpose_f16(const float* __restrict__ src, _Float16* __restrict__ dst,
                              int Kd, int Nsrc, int Npad)
{
  int idx = blockIdx.x * blockDim.x + threadIdx.x;
  if (idx >= Npad * Kd) return;
  int n = idx / Kd, kk = idx - n * Kd;
  dst[idx] = (n < Nsrc) ? (_Float16)src[(size_t)kk * Nsrc + n] : (_Float16)0.f;
}

// combined point-projection weight: rows 0..23 Wpq^T, 24..47 Wpv^T ; also bias concat
__global__ void build_wpp(const float* __restrict__ Wpq, const float* __restrict__ Wpv,
                          const float* __restrict__ bpq, const float* __restrict__ bpv,
                          _Float16* __restrict__ WppT, float* __restrict__ bpp)
{
  int idx = blockIdx.x * blockDim.x + threadIdx.x;
  if (idx >= 48 * CSZ) return;
  int n = idx / CSZ, kk = idx - n * CSZ;
  float v = (n < 24) ? Wpq[(size_t)kk * 24 + n] : Wpv[(size_t)kk * 24 + (n - 24)];
  WppT[idx] = (_Float16)v;
  if (idx < 48) bpp[idx] = (idx < 24) ? bpq[idx] : bpv[idx - 24];
}

// ---------------- layernorm + f16 conversion of s and sn ----------------
__global__ __launch_bounds__(128) void ln_kernel(const float* __restrict__ s,
    const float* __restrict__ lng, const float* __restrict__ lnb,
    _Float16* __restrict__ s16, _Float16* __restrict__ sn16)
{
  __shared__ float red[128];
  int t = blockIdx.x, tid = threadIdx.x;
  const float* row = s + (size_t)t * CSZ;
  float x0 = row[tid], x1 = row[tid + 128], x2 = row[tid + 256];
  red[tid] = x0 + x1 + x2; __syncthreads();
  for (int off = 64; off > 0; off >>= 1) { if (tid < off) red[tid] += red[tid + off]; __syncthreads(); }
  float mean = red[0] / CSZ; __syncthreads();
  float d0 = x0 - mean, d1 = x1 - mean, d2 = x2 - mean;
  red[tid] = d0 * d0 + d1 * d1 + d2 * d2; __syncthreads();
  for (int off = 64; off > 0; off >>= 1) { if (tid < off) red[tid] += red[tid + off]; __syncthreads(); }
  float inv = rsqrtf(red[0] / CSZ + 1e-5f);
  size_t base = (size_t)t * CSZ;
  s16[base + tid]       = (_Float16)x0;
  s16[base + tid + 128] = (_Float16)x1;
  s16[base + tid + 256] = (_Float16)x2;
  sn16[base + tid]       = (_Float16)(d0 * inv * lng[tid]       + lnb[tid]);
  sn16[base + tid + 128] = (_Float16)(d1 * inv * lng[tid + 128] + lnb[tid + 128]);
  sn16[base + tid + 256] = (_Float16)(d2 * inv * lng[tid + 256] + lnb[tid + 256]);
}

// ---------------- build transposed attention value matrix cvT (B,H,48,N) f16 -------
// cols 0..15 = v[j,h,c]; 16..39 = global points vg = R_j * kv_pts + trans*SCALE; 40..47 = 0
__global__ void build_cv(const float* __restrict__ kv, const float* __restrict__ kvpts,
                         const float* __restrict__ rot, const float* __restrict__ trans,
                         _Float16* __restrict__ cvT)
{
  int idx = blockIdx.x * blockDim.x + threadIdx.x;
  if (idx >= TOK * HH) return;
  int h = idx % HH, bn = idx / HH;
  int b = bn >> 9, j = bn & 511;
  const float* R = rot + (size_t)bn * 9;
  float t0 = trans[bn*3+0]*SCL_, t1 = trans[bn*3+1]*SCL_, t2 = trans[bn*3+2]*SCL_;
  _Float16* dst = cvT + ((size_t)(b * HH + h) * 48) * NN + j;
  const float* kvrow = kv + (size_t)bn * (2*HH*CC) + h * (2*CC);
#pragma unroll
  for (int c = 0; c < CC; ++c) dst[(size_t)c * NN] = (_Float16)kvrow[CC + c];
  const float* kp = kvpts + (size_t)bn * (HH*PVV*3) + h * (PVV*3);
#pragma unroll
  for (int p = 0; p < PVV; ++p) {
    float x = kp[p*3], y = kp[p*3+1], z = kp[p*3+2];
    dst[(size_t)(16 + p*3 + 0) * NN] = (_Float16)(R[0]*x + R[1]*y + R[2]*z + t0);
    dst[(size_t)(16 + p*3 + 1) * NN] = (_Float16)(R[3]*x + R[4]*y + R[5]*z + t1);
    dst[(size_t)(16 + p*3 + 2) * NN] = (_Float16)(R[6]*x + R[7]*y + R[8]*z + t2);
  }
#pragma unroll
  for (int c = 40; c < 48; ++c) dst[(size_t)c * NN] = (_Float16)0.f;
}

// ---------------- fused pair kernel: geometry -> gauss -> 3x WMMA GEMM -> logits ----
#define SM_R0   69632                     // gaussA 256x136 f16 (later aliased by hBuf 256x72)
#define SM_R1   36864                     // gBuf   256x72 f16 (later aliased by vfnBuf 256x16 f32)
#define SM_TAB  (4*(128+16+16+12))
#define SM_PAIR (SM_R0 + SM_R1 + SM_TAB)

__global__ __launch_bounds__(256) void pair_kernel(
    const float* __restrict__ rot, const float* __restrict__ trans,
    const float* __restrict__ mask,
    const float* __restrict__ qbuf, const float* __restrict__ kvbuf,
    const float* __restrict__ ppbuf,
    const float* __restrict__ Wvl, const float* __restrict__ means,
    const float* __restrict__ stds, const float* __restrict__ hws,
    const _Float16* __restrict__ WgT, const float* __restrict__ bg,
    const _Float16* __restrict__ W1T, const float* __restrict__ b1,
    const _Float16* __restrict__ W2T, const float* __restrict__ b2,
    float* __restrict__ gOut, float* __restrict__ aPre)
{
  extern __shared__ char smem[];
  _Float16* gaussA = (_Float16*)smem;                 // 256 x 136
  _Float16* hBuf   = (_Float16*)smem;                 // 256 x 72 (alias, after GEMM1)
  _Float16* gBuf   = (_Float16*)(smem + SM_R0);       // 256 x 72
  float*    vfnBuf = (float*)(smem + SM_R0);          // 256 x 16 (alias, after GEMM2)
  float*    sWvl   = (float*)(smem + SM_R0 + SM_R1);  // 128
  float*    sMean  = sWvl + 128;
  float*    sStd   = sMean + 16;
  float*    sHw    = sStd + 16;

  int tid = threadIdx.x, lane = tid & 31, w = tid >> 5;
  int tile = blockIdx.x;
  int b = tile >> 10, rem = tile & 1023;
  int i0 = (rem >> 5) << 4, j0 = (rem & 31) << 4;

  if (tid < 128) sWvl[tid] = Wvl[tid];
  if (tid < 16) { sMean[tid] = means[tid]; sStd[tid] = stds[tid]; }
  if (tid < 12) sHw[tid] = log1pf(expf(hws[tid])) * sqrtf(1.0f / (4.0f * 4.5f)) * (-0.5f);

  int il = tid >> 4, jl = tid & 15;
  int i = i0 + il, j = j0 + jl;
  int bi = b * NN + i, bj = b * NN + j;
  int m = tid;

  // ---- per-pair geometry ----
  float ri[9], rj[9];
#pragma unroll
  for (int x = 0; x < 9; ++x) { ri[x] = rot[(size_t)bi*9 + x]; rj[x] = rot[(size_t)bj*9 + x]; }
  float d0 = trans[bj*3+0] - trans[bi*3+0];
  float d1 = trans[bj*3+1] - trans[bi*3+1];
  float d2 = trans[bj*3+2] - trans[bi*3+2];
  float Rr[9], tr[3];
#pragma unroll
  for (int x = 0; x < 3; ++x)
#pragma unroll
    for (int z = 0; z < 3; ++z)
      Rr[x*3+z] = ri[x]*rj[z] + ri[3+x]*rj[3+z] + ri[6+x]*rj[6+z];
  tr[0] = ri[0]*d0 + ri[3]*d1 + ri[6]*d2;
  tr[1] = ri[1]*d0 + ri[4]*d1 + ri[7]*d2;
  tr[2] = ri[2]*d0 + ri[5]*d1 + ri[8]*d2;
  float pair_dist = (d0*d0 + d1*d1 + d2*d2) * (SCL_*SCL_);

  float vin[16][3];
  const float* vp = ppbuf + (size_t)bj * 48 + 24;
#pragma unroll
  for (int p = 0; p < 8; ++p) {
    float x = vp[p*3], y = vp[p*3+1], z = vp[p*3+2];
    vin[p][0] = Rr[0]*x + Rr[1]*y + Rr[2]*z + tr[0];
    vin[p][1] = Rr[3]*x + Rr[4]*y + Rr[5]*z + tr[1];
    vin[p][2] = Rr[6]*x + Rr[7]*y + Rr[8]*z + tr[2];
  }
  const float* qp = ppbuf + (size_t)bi * 48;
#pragma unroll
  for (int p = 0; p < 8; ++p) {
    vin[8+p][0] = qp[p*3]; vin[8+p][1] = qp[p*3+1]; vin[8+p][2] = qp[p*3+2];
  }
  __syncthreads();   // tables ready

#pragma unroll
  for (int o = 0; o < 8; ++o) {
    float ax = vin[o][0], ay = vin[o][1], az = vin[o][2];
#pragma unroll
    for (int p = 0; p < 16; ++p) {
      float ww = sWvl[o*16+p];
      ax += ww*vin[p][0]; ay += ww*vin[p][1]; az += ww*vin[p][2];
    }
    float vl = sqrtf(ax*ax + ay*ay + az*az + EPS_);
#pragma unroll
    for (int kk = 0; kk < 16; ++kk) {
      float zz = (vl - sMean[kk]) / sStd[kk];
      gaussA[m*136 + o*16 + kk] = (_Float16)expf(-0.5f * zz * zz);
    }
  }
  __syncthreads();

  int colbase = lane & 15, half_ = lane >> 4;

  // ---- GEMM1: g = gauss(256x128) @ Wg(128x64) + bg ----
#pragma unroll
  for (int mi = 0; mi < 2; ++mi) {
    int mt = (w << 1) + mi;
    int arow = (mt << 4) + colbase;
    v8f acc[4] = {};
    for (int k0 = 0; k0 < 128; k0 += 32) {
      v16h a = load_a_frag(gaussA, 136, arow, k0, lane);
#pragma unroll
      for (int nt = 0; nt < 4; ++nt) {
        v16h bf = load_b_frag(WgT, 128, (nt << 4) + colbase, k0, lane);
        acc[nt] = __builtin_amdgcn_wmma_f32_16x16x32_f16(false, a, false, bf, (short)0, acc[nt], false, false);
      }
    }
#pragma unroll
    for (int nt = 0; nt < 4; ++nt) {
      int col = (nt << 4) + colbase;
      float bv = bg[col];
#pragma unroll
      for (int r = 0; r < 8; ++r) {
        int mm = (mt << 4) + r + (half_ << 3);
        float gval = acc[nt][r] + bv;
        int iL = mm >> 4, jL = mm & 15;
        gOut[((((size_t)b * NN) + (i0 + iL)) * NN + (j0 + jL)) * GG + col] = gval;
        gBuf[mm * 72 + col] = (_Float16)gval;
      }
    }
  }
  __syncthreads();

  // ---- GEMM2: h1 = relu(g(256x64) @ W1(64x64) + b1) ----
#pragma unroll
  for (int mi = 0; mi < 2; ++mi) {
    int mt = (w << 1) + mi;
    int arow = (mt << 4) + colbase;
    v8f acc[4] = {};
    for (int k0 = 0; k0 < 64; k0 += 32) {
      v16h a = load_a_frag(gBuf, 72, arow, k0, lane);
#pragma unroll
      for (int nt = 0; nt < 4; ++nt) {
        v16h bf = load_b_frag(W1T, 64, (nt << 4) + colbase, k0, lane);
        acc[nt] = __builtin_amdgcn_wmma_f32_16x16x32_f16(false, a, false, bf, (short)0, acc[nt], false, false);
      }
    }
#pragma unroll
    for (int nt = 0; nt < 4; ++nt) {
      int col = (nt << 4) + colbase;
      float bv = b1[col];
#pragma unroll
      for (int r = 0; r < 8; ++r) {
        int mm = (mt << 4) + r + (half_ << 3);
        hBuf[mm * 72 + col] = (_Float16)fmaxf(acc[nt][r] + bv, 0.f);
      }
    }
  }
  __syncthreads();

  // ---- GEMM3: vfn = h1(256x64) @ W2(64x16pad) + b2 ----
#pragma unroll
  for (int mi = 0; mi < 2; ++mi) {
    int mt = (w << 1) + mi;
    int arow = (mt << 4) + colbase;
    v8f acc = {};
    for (int k0 = 0; k0 < 64; k0 += 32) {
      v16h a  = load_a_frag(hBuf, 72, arow, k0, lane);
      v16h bf = load_b_frag(W2T, 64, colbase, k0, lane);
      acc = __builtin_amdgcn_wmma_f32_16x16x32_f16(false, a, false, bf, (short)0, acc, false, false);
    }
    if (colbase < 12) {
      float bv = b2[colbase];
#pragma unroll
      for (int r = 0; r < 8; ++r) {
        int mm = (mt << 4) + r + (half_ << 3);
        vfnBuf[mm * 16 + colbase] = acc[r] + bv;
      }
    }
  }
  __syncthreads();

  // ---- combine: pre-softmax logits ----
  float maskterm = INF_ * (mask[bi] * mask[bj] - 1.0f);
  const float* qrow = qbuf + (size_t)bi * (HH*CC);
  const float* krow = kvbuf + (size_t)bj * (2*HH*CC);
#pragma unroll
  for (int h = 0; h < HH; ++h) {
    float dot = 0.f;
#pragma unroll
    for (int c = 0; c < CC; ++c) dot += qrow[h*16+c] * krow[h*32+c];
    float val = dot * 0.125f + vfnBuf[m*16+h] * 0.5f + pair_dist * sHw[h] + maskterm;
    aPre[(((size_t)(b*HH + h)) * NN + i) * NN + j] = val;
  }
}

// ---------------- row softmax -> f16 ----------------
__global__ __launch_bounds__(256) void softmax_rows(const float* __restrict__ aPre,
                                                    _Float16* __restrict__ a16)
{
  __shared__ float red[256];
  int row = blockIdx.x, tid = threadIdx.x;
  const float* p = aPre + (size_t)row * NN;
  float x0 = p[tid], x1 = p[tid + 256];
  red[tid] = fmaxf(x0, x1); __syncthreads();
  for (int off = 128; off > 0; off >>= 1) { if (tid < off) red[tid] = fmaxf(red[tid], red[tid + off]); __syncthreads(); }
  float mx = red[0]; __syncthreads();
  float e0 = expf(x0 - mx), e1 = expf(x1 - mx);
  red[tid] = e0 + e1; __syncthreads();
  for (int off = 128; off > 0; off >>= 1) { if (tid < off) red[tid] += red[tid + off]; __syncthreads(); }
  float inv = 1.f / red[0];
  _Float16* q = a16 + (size_t)row * NN;
  q[tid] = (_Float16)(e0 * inv); q[tid + 256] = (_Float16)(e1 * inv);
}

// ---------------- apply attention: OV(B,H,N,48) = a(N,N) @ cv(N,48) per (b,h) -------
__global__ __launch_bounds__(256) void attn_apply(const _Float16* __restrict__ a16,
                                                  const _Float16* __restrict__ cvT,
                                                  float* __restrict__ OV)
{
  int bh = blockIdx.y;
  int lane = threadIdx.x & 31;
  int wid = blockIdx.x * 8 + (threadIdx.x >> 5);   // 0..95
  int mt = wid / 3, nt = wid - mt * 3;
  int row = (mt << 4) + (lane & 15);
  int col = (nt << 4) + (lane & 15);
  const _Float16* A  = a16 + (size_t)bh * NN * NN;
  const _Float16* Bt = cvT + (size_t)bh * 48 * NN;
  v8f acc = {};
  for (int k0 = 0; k0 < NN; k0 += 32) {
    v16h a = load_a_frag(A,  NN, row, k0, lane);
    v16h b = load_b_frag(Bt, NN, col, k0, lane);
    acc = __builtin_amdgcn_wmma_f32_16x16x32_f16(false, a, false, b, (short)0, acc, false, false);
  }
  int half_ = lane >> 4;
#pragma unroll
  for (int r = 0; r < 8; ++r) {
    int mm = (mt << 4) + r + (half_ << 3);
    OV[((size_t)bh * NN + mm) * 48 + col] = acc[r];
  }
}

// ---------------- feats: local-frame rotation, dists, concat (f16) ------------------
__global__ void build_feats(const float* __restrict__ OV, const float* __restrict__ rot,
                            const float* __restrict__ trans, _Float16* __restrict__ feats)
{
  int t = blockIdx.x * blockDim.x + threadIdx.x;  // token
  if (t >= TOK) return;
  int b = t >> 9, i = t & 511;
  const float* R = rot + (size_t)t * 9;
  float t0 = trans[t*3+0]*SCL_, t1 = trans[t*3+1]*SCL_, t2 = trans[t*3+2]*SCL_;
  _Float16* f = feats + (size_t)t * 576;
  for (int h = 0; h < HH; ++h) {
    const float* ov = OV + ((size_t)(b * HH + h) * NN + i) * 48;
#pragma unroll
    for (int c = 0; c < CC; ++c) f[h*16 + c] = (_Float16)ov[c];
#pragma unroll
    for (int p = 0; p < PVV; ++p) {
      float e0 = ov[16 + p*3 + 0] - t0;
      float e1 = ov[16 + p*3 + 1] - t1;
      float e2 = ov[16 + p*3 + 2] - t2;
      float lx = R[0]*e0 + R[3]*e1 + R[6]*e2;   // rot^T
      float ly = R[1]*e0 + R[4]*e1 + R[7]*e2;
      float lz = R[2]*e0 + R[5]*e1 + R[8]*e2;
      int kp = h * PVV + p;
      f[192 +   0 + kp] = (_Float16)lx;
      f[192 +  96 + kp] = (_Float16)ly;
      f[192 + 192 + kp] = (_Float16)lz;
      f[192 + 288 + kp] = (_Float16)sqrtf(lx*lx + ly*ly + lz*lz + EPS_);
    }
  }
}

// ---------------- host launcher ----------------
extern "C" void kernel_launch(void* const* d_in, const int* in_sizes, int n_in,
                              void* d_out, int out_size, void* d_ws, size_t ws_size,
                              hipStream_t stream) {
  const float* s     = (const float*)d_in[0];
  const float* rot   = (const float*)d_in[1];
  const float* trans = (const float*)d_in[2];
  const float* mask  = (const float*)d_in[3];
  const float* Wq    = (const float*)d_in[4];
  const float* bq    = (const float*)d_in[5];
  const float* Wkv   = (const float*)d_in[6];
  const float* bkv   = (const float*)d_in[7];
  const float* hw    = (const float*)d_in[8];
  const float* lng   = (const float*)d_in[9];
  const float* lnb   = (const float*)d_in[10];
  const float* Wpq   = (const float*)d_in[11];
  const float* bpq   = (const float*)d_in[12];
  const float* Wpv   = (const float*)d_in[13];
  const float* bpv   = (const float*)d_in[14];
  const float* Wvl   = (const float*)d_in[15];
  const float* gbm   = (const float*)d_in[16];
  const float* gbs   = (const float*)d_in[17];
  const float* Wg    = (const float*)d_in[18];
  const float* bg    = (const float*)d_in[19];
  const float* W1    = (const float*)d_in[20];
  const float* b1    = (const float*)d_in[21];
  const float* W2    = (const float*)d_in[22];
  const float* b2    = (const float*)d_in[23];
  const float* Wkvp  = (const float*)d_in[24];
  const float* bkvp  = (const float*)d_in[25];
  const float* Wout  = (const float*)d_in[26];
  const float* bout  = (const float*)d_in[27];

  char* w = (char*)d_ws;
  // workspace layout (bytes)
  _Float16* s16    = (_Float16*)(w + 0);          //  786432
  _Float16* sn16   = (_Float16*)(w + 786432);     //  786432
  _Float16* WqT    = (_Float16*)(w + 1572864);    //  147456
  _Float16* WkvT   = (_Float16*)(w + 1720320);    //  294912
  _Float16* WkvpT  = (_Float16*)(w + 2015232);    //  221184
  _Float16* WppT   = (_Float16*)(w + 2236416);    //   36864
  float*    bpp    = (float*)   (w + 2273280);    //     256
  _Float16* WgT    = (_Float16*)(w + 2273536);    //   16384
  _Float16* W1T    = (_Float16*)(w + 2289920);    //    8192
  _Float16* W2T    = (_Float16*)(w + 2298112);    //    2048
  _Float16* WoutT  = (_Float16*)(w + 2300160);    //  442368
  float*    Qb     = (float*)   (w + 2742528);    //  786432
  float*    KVb    = (float*)   (w + 3528960);    // 1572864
  float*    PPb    = (float*)   (w + 5101824);    //  196608
  float*    KVPTS  = (float*)   (w + 5298432);    // 1179648
  _Float16* cvT    = (_Float16*)(w + 6478080);    // 1179648
  float*    aPre   = (float*)   (w + 7657728);    // 25165824
  _Float16* a16    = (_Float16*)(w + 32823552);   // 12582912
  float*    OV     = (float*)   (w + 45406464);   // 2359296
  _Float16* feats  = (_Float16*)(w + 47765760);   // 1179648

  float* outS = (float*)d_out;                    // s_out: 1024*384
  float* outG = (float*)d_out + (size_t)TOK * CSZ; // g: B*N*N*64

  hipFuncSetAttribute((const void*)pair_kernel,
                      hipFuncAttributeMaxDynamicSharedMemorySize, SM_PAIR);

  // weight conversions / transposes
  auto tb = [](int n) { return (n + 255) / 256; };
  transpose_f16<<<tb(CSZ*192), 256, 0, stream>>>(Wq,   WqT,   CSZ, 192, 192);
  transpose_f16<<<tb(CSZ*384), 256, 0, stream>>>(Wkv,  WkvT,  CSZ, 384, 384);
  transpose_f16<<<tb(CSZ*288), 256, 0, stream>>>(Wkvp, WkvpT, CSZ, 288, 288);
  transpose_f16<<<tb(128*64),  256, 0, stream>>>(Wg,   WgT,   128, 64, 64);
  transpose_f16<<<tb(64*64),   256, 0, stream>>>(W1,   W1T,   64, 64, 64);
  transpose_f16<<<tb(64*16),   256, 0, stream>>>(W2,   W2T,   64, 12, 16);
  transpose_f16<<<tb(576*384), 256, 0, stream>>>(Wout, WoutT, 576, 384, 384);
  build_wpp<<<tb(48*CSZ), 256, 0, stream>>>(Wpq, Wpv, bpq, bpv, WppT, bpp);

  // layernorm + f16 activations
  ln_kernel<<<TOK, 128, 0, stream>>>(s, lng, lnb, s16, sn16);

  // projections (WMMA GEMMs)
  auto gblk = [](int M, int Nc) { return (((M >> 4) * (Nc >> 4)) + 7) / 8; };
  gemm_bias_f16<<<gblk(TOK,192), 256, 0, stream>>>(s16,  WqT,   bq,   Qb,    TOK, 192, CSZ);
  gemm_bias_f16<<<gblk(TOK,384), 256, 0, stream>>>(s16,  WkvT,  bkv,  KVb,   TOK, 384, CSZ);
  gemm_bias_f16<<<gblk(TOK,288), 256, 0, stream>>>(s16,  WkvpT, bkvp, KVPTS, TOK, 288, CSZ);
  gemm_bias_f16<<<gblk(TOK,48),  256, 0, stream>>>(sn16, WppT,  bpp,  PPb,   TOK, 48,  CSZ);

  // transposed attention-value matrix
  build_cv<<<tb(TOK*HH), 256, 0, stream>>>(KVb, KVPTS, rot, trans, cvT);

  // fused pairwise kernel (geometry + gaussians + 3 WMMA GEMMs + logits + g output)
  pair_kernel<<<BB * 32 * 32, 256, SM_PAIR, stream>>>(
      rot, trans, mask, Qb, KVb, PPb, Wvl, gbm, gbs, hw,
      WgT, bg, W1T, b1, W2T, b2, outG, aPre);

  // softmax rows (B*H*N)
  softmax_rows<<<BB * HH * NN, 256, 0, stream>>>(aPre, a16);

  // apply attention per (b,h)
  attn_apply<<<dim3(12, BB * HH), 256, 0, stream>>>(a16, cvT, OV);

  // local-frame features + final projection
  build_feats<<<tb(TOK), 256, 0, stream>>>(OV, rot, trans, feats);
  gemm_bias_f16<<<gblk(TOK,384), 256, 0, stream>>>(feats, WoutT, bout, outS, TOK, 384, 576);

  (void)in_sizes; (void)n_in; (void)out_size; (void)ws_size;
}
